// SceneSayerODE_86973087744258
// MI455X (gfx1250) — matrix operand order repaired and checked by reading, not compiled, not run on hardware
//
#include <hip/hip_runtime.h>
#include <hip/hip_bf16.h>
#include <math.h>

// ---------------------------------------------------------------------------
// Reference: N=2048 rows, D=1936 latent, H=2048 hidden; 30 frames x 2 substeps.
// Internally D is padded -> 2048 (GDIM) with zeroed weight rows/cols so both
// GEMMs are exact and compile-time sized 2048x2048x2048 (no bounds checks).
// ---------------------------------------------------------------------------
#define NROWS 2048
#define DDIM  1936
#define GDIM  2048          // padded latent == hidden == row count
#define TFRAMES 30
#define SUBSTEPS 2
#define NHEAD 26

typedef __bf16 bf16;
typedef bf16  bf16x4  __attribute__((ext_vector_type(4)));
typedef bf16  bf16x8  __attribute__((ext_vector_type(8)));
typedef bf16  v16bf   __attribute__((ext_vector_type(16)));
typedef float v8f     __attribute__((ext_vector_type(8)));

#define TILE 128
#define LDSK 40             // 32 k + 8 pad (bank stagger), rows stay 16B aligned
#define KTILES (GDIM / 32)  // 64

// ---------------------------------------------------------------------------
// GEMM (compile-time 2048x2048x2048):
//   Out = op( A[2048x2048] * Wt[2048x2048]^T + bias )
//   A_BF16: A stored bf16 (else f32, converted while staging)
//   TANH_OUT: epilogue tanh + bf16 store (else f32 store)
// 256 threads = 8 waves; block tile 128x128; wave tile 64x32 = 4x2 WMMA frags.
// Double-buffered LDS, register-pipelined global loads:
//   barrier; gload(kt+1); compute buf[kt&1]; commit buf[(kt+1)&1]
// -> one barrier per k-step, global latency hidden behind the 8-WMMA block.
// ---------------------------------------------------------------------------
template <bool A_BF16, bool TANH_OUT>
__global__ __launch_bounds__(256)
void gemm2048(const void* __restrict__ Ap, const bf16* __restrict__ Wt,
              const float* __restrict__ bias, void* __restrict__ Outp) {
  __shared__ __align__(16) bf16 As[2][TILE][LDSK];
  __shared__ __align__(16) bf16 Bs[2][TILE][LDSK];

  const int tid   = threadIdx.x;
  const int tileN = blockIdx.x;   // output-column tile
  const int tileM = blockIdx.y;   // row tile
  const int wave  = tid >> 5, lane = tid & 31;
  const int wm = wave >> 2, wn = wave & 3;    // 2 x 4 wave grid
  const int lm = lane & 15, lh = lane >> 4;   // lane row, lane half

  v8f acc[4][2];
#pragma unroll
  for (int i = 0; i < 4; ++i)
#pragma unroll
    for (int j = 0; j < 2; ++j)
#pragma unroll
      for (int e = 0; e < 8; ++e) acc[i][j][e] = 0.0f;

  // staging-thread mapping (compile-time):
  //  f32 A : 4 chunks of float4   -> row = ch>>3, kq = (ch&7)*4
  //  bf16 A: 2 chunks of bf16x8   -> row = ch>>2, kq = (ch&3)*8
  //  B     : 2 chunks of bf16x8   -> same as bf16 A
  const float* Af = (const float*)Ap;
  const bf16*  Ab = (const bf16*)Ap;

  float4 aF[4];
  bf16x8 aB[2];
  bf16x8 bR[2];

  auto load_tile = [&](int k0) {
    if (A_BF16) {
#pragma unroll
      for (int c = 0; c < 2; ++c) {
        const int ch = tid + (c << 8);
        const int row = ch >> 2, kq = (ch & 3) << 3;
        aB[c] = *(const bf16x8*)(Ab + (size_t)(tileM * TILE + row) * GDIM + k0 + kq);
      }
    } else {
#pragma unroll
      for (int c = 0; c < 4; ++c) {
        const int ch = tid + (c << 8);
        const int row = ch >> 3, kq = (ch & 7) << 2;
        aF[c] = *(const float4*)(Af + (size_t)(tileM * TILE + row) * GDIM + k0 + kq);
      }
    }
#pragma unroll
    for (int c = 0; c < 2; ++c) {
      const int ch = tid + (c << 8);
      const int n = ch >> 2, kq = (ch & 3) << 3;
      bR[c] = *(const bf16x8*)(Wt + (size_t)(tileN * TILE + n) * GDIM + k0 + kq);
    }
  };

  auto commit_tile = [&](int buf) {
    if (A_BF16) {
#pragma unroll
      for (int c = 0; c < 2; ++c) {
        const int ch = tid + (c << 8);
        const int row = ch >> 2, kq = (ch & 3) << 3;
        *(bf16x8*)&As[buf][row][kq] = aB[c];
      }
    } else {
#pragma unroll
      for (int c = 0; c < 4; ++c) {
        const int ch = tid + (c << 8);
        const int row = ch >> 3, kq = (ch & 7) << 2;
        bf16x4 v;
        v[0] = (bf16)aF[c].x; v[1] = (bf16)aF[c].y;
        v[2] = (bf16)aF[c].z; v[3] = (bf16)aF[c].w;
        *(bf16x4*)&As[buf][row][kq] = v;
      }
    }
#pragma unroll
    for (int c = 0; c < 2; ++c) {
      const int ch = tid + (c << 8);
      const int n = ch >> 2, kq = (ch & 3) << 3;
      *(bf16x8*)&Bs[buf][n][kq] = bR[c];
    }
  };

  // prologue: tile 0 into buffer 0
  load_tile(0);
  commit_tile(0);

  for (int kt = 0; kt < KTILES; ++kt) {
    __syncthreads();            // buf[kt&1] committed; reads of buf[(kt+1)&1] (iter kt-1) done
    const int buf = kt & 1;

    if (kt + 1 < KTILES) load_tile((kt + 1) << 5);   // in flight during WMMAs

    // ---- fragments per ISA wave32 layout + 8 WMMAs ----
    v16bf afrag[4], bfrag[2];
#pragma unroll
    for (int i = 0; i < 4; ++i) {
      const int m = wm * 64 + i * 16 + lm;
      const bf16x8 lo = *(const bf16x8*)&As[buf][m][lh * 8];       // K: kb .. kb+7
      const bf16x8 hi = *(const bf16x8*)&As[buf][m][lh * 8 + 16];  // K: kb+16 .. kb+23
      afrag[i] = __builtin_shufflevector(lo, hi, 0, 1, 2, 3, 4, 5, 6, 7,
                                         8, 9, 10, 11, 12, 13, 14, 15);
    }
#pragma unroll
    for (int j = 0; j < 2; ++j) {
      const int n = wn * 32 + j * 16 + lm;
      const bf16x8 lo = *(const bf16x8*)&Bs[buf][n][lh * 8];       // K: kb .. kb+7
      const bf16x8 hi = *(const bf16x8*)&Bs[buf][n][lh * 8 + 16];  // K: kb+16 .. kb+23
      bfrag[j] = __builtin_shufflevector(lo, hi, 0, 1, 2, 3, 4, 5, 6, 7,
                                         8, 9, 10, 11, 12, 13, 14, 15);
    }
#pragma unroll
    for (int i = 0; i < 4; ++i)
#pragma unroll
      for (int j = 0; j < 2; ++j)
        acc[i][j] = __builtin_amdgcn_wmma_f32_16x16x32_bf16(
            false, afrag[i], false, bfrag[j], (short)0, acc[i][j], false, false);

    if (kt + 1 < KTILES) commit_tile((kt + 1) & 1);  // after compute: wait_loadcnt lands here
  }

  // ---- epilogue: +bias, optional tanh (C/D layout: elem r -> row r + 8*lh) ----
#pragma unroll
  for (int j = 0; j < 2; ++j) {
    const int col = tileN * TILE + wn * 32 + j * 16 + lm;
    const float bv = bias[col];
#pragma unroll
    for (int i = 0; i < 4; ++i) {
      const int rowBase = tileM * TILE + wm * 64 + i * 16 + lh * 8;
#pragma unroll
      for (int r = 0; r < 8; ++r) {
        const float v = acc[i][j][r] + bv;
        const size_t idx = (size_t)(rowBase + r) * GDIM + col;
        if (TANH_OUT) ((bf16*)Outp)[idx] = (bf16)tanhf(v);
        else          ((float*)Outp)[idx] = v;
      }
    }
  }
}

// ---------------------------------------------------------------------------
// RK4 stage combine over padded [2048 x 2048] state (float4-vectorized).
//  s=1: kacc=k;   ztmp=z+0.5dt*k    s=2: kacc+=2k; ztmp=z+0.5dt*k
//  s=3: kacc+=2k; ztmp=z+dt*k       s=4: z += dt/6*(kacc+k)
// Pad columns stay exactly zero (weights/bias pads are zero).
// ---------------------------------------------------------------------------
__global__ __launch_bounds__(256)
void rk4_combine(float* __restrict__ zbase, float* __restrict__ ztmp,
                 float* __restrict__ kacc, const float* __restrict__ knew,
                 int stage, float dt) {
  const int i = blockIdx.x * blockDim.x + threadIdx.x;
  const float4 k = ((const float4*)knew)[i];
  if (stage == 4) {
    const float4 ka = ((const float4*)kacc)[i];
    float4 z = ((const float4*)zbase)[i];
    const float w = dt * (1.0f / 6.0f);
    z.x += w * (ka.x + k.x); z.y += w * (ka.y + k.y);
    z.z += w * (ka.z + k.z); z.w += w * (ka.w + k.w);
    ((float4*)zbase)[i] = z;
  } else {
    const float4 z = ((const float4*)zbase)[i];
    float4 ka;
    if (stage == 1) { ka = k; }
    else {
      ka = ((const float4*)kacc)[i];
      ka.x += 2.0f * k.x; ka.y += 2.0f * k.y; ka.z += 2.0f * k.z; ka.w += 2.0f * k.w;
    }
    ((float4*)kacc)[i] = ka;
    const float c = (stage == 3) ? dt : 0.5f * dt;
    float4 zt;
    zt.x = z.x + c * k.x; zt.y = z.y + c * k.y;
    zt.z = z.z + c * k.z; zt.w = z.w + c * k.w;
    ((float4*)ztmp)[i] = zt;
  }
}

// ---------------------------------------------------------------------------
// Heads: out[row][c] = dot(z[row, 0:1936], Wcat[:,c]) + bcat[c]
// One block per row (z row stride GDIM); 4 waves cover 26 cols; shfl reduce.
// ---------------------------------------------------------------------------
__global__ __launch_bounds__(128)
void heads_kernel(const float* __restrict__ z, const float* __restrict__ Wcat,
                  const float* __restrict__ bcat, float* __restrict__ out) {
  __shared__ float zrow[DDIM];
  const int row = blockIdx.x;
  const float* zr = z + (size_t)row * GDIM;
  for (int i = threadIdx.x; i < DDIM; i += blockDim.x) zrow[i] = zr[i];
  __syncthreads();
  const int wave = threadIdx.x >> 5, lane = threadIdx.x & 31;
  for (int c = wave; c < NHEAD; c += 4) {
    float s = 0.0f;
    for (int k = lane; k < DDIM; k += 32) s += zrow[k] * Wcat[(size_t)k * NHEAD + c];
    for (int off = 16; off > 0; off >>= 1) s += __shfl_xor(s, off, 32);
    if (lane == 0) out[(size_t)row * NHEAD + c] = s + bcat[c];
  }
}

// ---------------------------------------------------------------------------
// Prep kernels (one-time): padded transposed bf16 weights, padded bias,
// padded z0 copy, heads concat pack.
// ---------------------------------------------------------------------------
__global__ void build_w1t(const float* __restrict__ W1, bf16* __restrict__ W1t) {
  // W1 [1936][2048] f32 -> W1t [2048][2048] bf16, W1t[h][d] = d<1936 ? W1[d][h] : 0
  const int i = blockIdx.x * blockDim.x + threadIdx.x;   // i = h*2048 + d
  const int h = i >> 11, d = i & (GDIM - 1);
  W1t[i] = (d < DDIM) ? (bf16)W1[(size_t)d * GDIM + h] : (bf16)0.0f;
}
__global__ void build_w2t(const float* __restrict__ W2, bf16* __restrict__ W2t) {
  // W2 [2048][1936] f32 -> W2t [2048][2048] bf16, W2t[m][k] = m<1936 ? W2[k][m] : 0
  const int i = blockIdx.x * blockDim.x + threadIdx.x;   // i = m*2048 + k
  const int m = i >> 11, k = i & (GDIM - 1);
  W2t[i] = (m < DDIM) ? (bf16)W2[(size_t)k * DDIM + m] : (bf16)0.0f;
}
__global__ void pad_bias(const float* __restrict__ b2, float* __restrict__ b2p) {
  const int i = blockIdx.x * blockDim.x + threadIdx.x;
  if (i < GDIM) b2p[i] = (i < DDIM) ? b2[i] : 0.0f;
}
__global__ void pad_copy_z0(const float* __restrict__ z0, float* __restrict__ zb) {
  const int i = blockIdx.x * blockDim.x + threadIdx.x;   // i = row*2048 + col
  const int row = i >> 11, col = i & (GDIM - 1);
  zb[i] = (col < DDIM) ? z0[(size_t)row * DDIM + col] : 0.0f;
}
__global__ void pack_heads(const float* __restrict__ Wa, const float* __restrict__ Wsp,
                           const float* __restrict__ Wc, const float* __restrict__ ba,
                           const float* __restrict__ bsp, const float* __restrict__ bc,
                           float* __restrict__ Wcat, float* __restrict__ bcat) {
  const int i = blockIdx.x * blockDim.x + threadIdx.x;
  if (i < DDIM * NHEAD) {
    const int d = i / NHEAD, c = i % NHEAD;
    float v;
    if (c < 3)      v = Wa[(size_t)d * 3 + c];
    else if (c < 9) v = Wsp[(size_t)d * 6 + (c - 3)];
    else            v = Wc[(size_t)d * 17 + (c - 9)];
    Wcat[i] = v;
  }
  if (i < NHEAD) bcat[i] = (i < 3) ? ba[i] : (i < 9) ? bsp[i - 3] : bc[i - 9];
}

// ---------------------------------------------------------------------------
extern "C" void kernel_launch(void* const* d_in, const int* in_sizes, int n_in,
                              void* d_out, int out_size, void* d_ws, size_t ws_size,
                              hipStream_t stream) {
  const float* z0  = (const float*)d_in[0];
  const float* W1  = (const float*)d_in[1];
  const float* b1  = (const float*)d_in[2];
  const float* W2  = (const float*)d_in[3];
  const float* b2  = (const float*)d_in[4];
  const float* Wa  = (const float*)d_in[5];
  const float* ba  = (const float*)d_in[6];
  const float* Wsp = (const float*)d_in[7];
  const float* bsp = (const float*)d_in[8];
  const float* Wc  = (const float*)d_in[9];
  const float* bc  = (const float*)d_in[10];
  float* out = (float*)d_out;

  const size_t NG = (size_t)NROWS * GDIM;   // padded state elements

  char* ws = (char*)d_ws;
  size_t off = 0;
  auto alloc = [&](size_t bytes) -> void* {
    void* p = ws + off;
    off = (off + bytes + 255) & ~(size_t)255;
    return p;
  };
  float* zbase = (float*)alloc(NG * 4);
  float* ztmp  = (float*)alloc(NG * 4);
  float* kacc  = (float*)alloc(NG * 4);
  float* knew  = (float*)alloc(NG * 4);
  bf16*  hbuf  = (bf16*)alloc(NG * 2);
  bf16*  W1t   = (bf16*)alloc((size_t)GDIM * GDIM * 2);
  bf16*  W2t   = (bf16*)alloc((size_t)GDIM * GDIM * 2);
  float* b2p   = (float*)alloc(GDIM * 4);
  float* Wcat  = (float*)alloc((size_t)DDIM * NHEAD * 4);
  float* bcat  = (float*)alloc(NHEAD * 4);

  // one-time prep (trivial vs ~8 TFLOP of GEMMs)
  {
    const int nw = GDIM * GDIM;
    build_w1t<<<nw / 256, 256, 0, stream>>>(W1, W1t);
    build_w2t<<<nw / 256, 256, 0, stream>>>(W2, W2t);
    pad_bias<<<(GDIM + 255) / 256, 256, 0, stream>>>(b2, b2p);
    pad_copy_z0<<<(int)(NG / 256), 256, 0, stream>>>(z0, zbase);
    const int nh = DDIM * NHEAD;
    pack_heads<<<(nh + 255) / 256, 256, 0, stream>>>(Wa, Wsp, Wc, ba, bsp, bc, Wcat, bcat);
  }

  const dim3 blk(256);
  const dim3 gg(GDIM / TILE, NROWS / TILE);     // (16,16)
  const dim3 gc((unsigned)(NG / 4 / 256));      // combine: float4 per thread
  const float dt = 1.0f / (float)SUBSTEPS;

  for (int f = 0; f < TFRAMES; ++f) {
    for (int s = 0; s < SUBSTEPS; ++s) {
      for (int stage = 1; stage <= 4; ++stage) {
        const float* zin = (stage == 1) ? zbase : ztmp;
        // h = tanh(zin @ W1 + b1)   [f32 A -> bf16+tanh out]
        gemm2048<false, true><<<gg, blk, 0, stream>>>(zin, W1t, b1, hbuf);
        // knew = h @ W2 + b2        [bf16 A -> f32 out]
        gemm2048<true, false><<<gg, blk, 0, stream>>>(hbuf, W2t, b2p, knew);
        rk4_combine<<<gc, blk, 0, stream>>>(zbase, ztmp, kacc, knew, stage, dt);
      }
    }
    heads_kernel<<<NROWS, 128, 0, stream>>>(zbase, Wcat, bcat,
                                            out + (size_t)f * NROWS * NHEAD);
  }
}
